// AttentionFake_11768210391710
// MI455X (gfx1250) — matrix operand order repaired and checked by reading, not compiled, not run on hardware
//
#include <hip/hip_runtime.h>
#include <hip/hip_bf16.h>
#include <math.h>

typedef __attribute__((ext_vector_type(16))) _Float16 v16h;
typedef __attribute__((ext_vector_type(8)))  _Float16 v8h;
typedef __attribute__((ext_vector_type(8)))  float    v8f;

#define NBATCH 8
#define SEQ    4096
#define DMODEL 128
#define NROWS  (NBATCH * SEQ)   // 32768

__device__ __forceinline__ v8f wmma_f16(v16h a, v16h b, v8f c) {
    // D = A(16x32 f16) * B(32x16 f16) + C(16x16 f32)
    return __builtin_amdgcn_wmma_f32_16x16x32_f16(
        /*neg_a=*/false, a, /*neg_b=*/false, b,
        /*c_mod=*/(short)0, c, /*reuse_a=*/false, /*reuse_b=*/false);
}

// ---------------------------------------------------------------------------
// Kernel 1: k/q/v projections.  x[f32] * W^T[f32] + b  ->  f16 outputs.
// k,q stored row-major [row][d]; v stored transposed per batch [b][d][n].
// One wave handles a 16-row tile; 4 waves per block.
// ---------------------------------------------------------------------------
__global__ __launch_bounds__(128, 1)
void proj_qkv_kernel(const float* __restrict__ x,
                     const float* __restrict__ Wk, const float* __restrict__ bk,
                     const float* __restrict__ Wq, const float* __restrict__ bq,
                     const float* __restrict__ Wv, const float* __restrict__ bv,
                     _Float16* __restrict__ kf,
                     _Float16* __restrict__ qf,
                     _Float16* __restrict__ vt)
{
    const int w       = threadIdx.x >> 5;
    const int lane    = threadIdx.x & 31;
    const int half    = lane >> 4;
    const int l16     = lane & 15;
    const int rowTile = blockIdx.x * 4 + w;       // 0..2047
    const int rowbase = rowTile * 16;

    // ---- A matrix: 16 x rows, 128 wide, f32 -> f16, WMMA A layout ----
    // lane(half,m): element e<8 -> K = 8*half + e ; e>=8 -> K = 16 + 8*half + (e-8)
    v16h a[4];
    {
        const float* xrow = x + (size_t)(rowbase + l16) * DMODEL;
        for (int c = 0; c < 4; ++c) {
            const int d0 = 32 * c + 8 * half;
            v16h av;
            for (int e = 0; e < 8; ++e) av[e]     = (_Float16)xrow[d0 + e];
            for (int e = 0; e < 8; ++e) av[8 + e] = (_Float16)xrow[d0 + 16 + e];
            a[c] = av;
        }
    }

    auto project = [&](const float* __restrict__ W, const float* __restrict__ bias,
                       _Float16* __restrict__ out_rm, _Float16* __restrict__ out_tr) {
        for (int ct = 0; ct < 8; ++ct) {             // output column tile, d = 16*ct..
            const int dbase = 16 * ct;
            v8f acc = {};
            const float* wrow = W + (size_t)(dbase + l16) * DMODEL; // B lane = out column
            for (int c = 0; c < 4; ++c) {
                const int f0 = 32 * c + 8 * half;    // B layout: e<8 -> K = 8*half + e ; e>=8 -> K = 16 + 8*half + (e-8)
                v16h bmat;
                for (int e = 0; e < 8; ++e) { bmat[e] = (_Float16)wrow[f0 + e]; bmat[8 + e] = (_Float16)wrow[f0 + 16 + e]; }
                acc = wmma_f16(a[c], bmat, acc);
            }
            const float bb = bias[dbase + l16];
            for (int r = 0; r < 8; ++r) acc[r] += bb;

            if (out_rm) {
                // C/D layout: lane(half,n) holds rows M = 8*half + r, col dbase+n
                for (int r = 0; r < 8; ++r) {
                    const int row = rowbase + 8 * half + r;
                    out_rm[(size_t)row * DMODEL + dbase + l16] = (_Float16)acc[r];
                }
            } else {
                // transposed store: vt[b][d][n]; consecutive r are contiguous -> b128
                v8h pk;
                for (int r = 0; r < 8; ++r) pk[r] = (_Float16)acc[r];
                const int b    = rowTile >> 8;              // rowbase / 4096
                const int nloc = (rowbase & (SEQ - 1)) + 8 * half;
                _Float16* dst = out_tr + ((size_t)b * DMODEL + dbase + l16) * SEQ + nloc;
                *(v8h*)dst = pk;
            }
        }
    };

    project(Wk, bk, kf, nullptr);
    project(Wq, bq, qf, nullptr);
    project(Wv, bv, nullptr, vt);
}

// ---------------------------------------------------------------------------
// Kernel 2: fused attention (flash-style, online softmax).
//   S = k_tile(16x128) @ q^T, softmax over q index, O += P @ v.
// One wave owns 16 output rows; accumulator 16x128 f32 lives in 8 v8f tiles.
// ---------------------------------------------------------------------------
__global__ __launch_bounds__(128, 1)
void attn_kernel(const _Float16* __restrict__ kf,
                 const _Float16* __restrict__ qf,
                 const _Float16* __restrict__ vt,
                 float* __restrict__ out)
{
    __shared__ _Float16 lds_p[4][16 * 32];   // per-wave P staging (C-layout -> A-layout)

    const int w       = threadIdx.x >> 5;
    const int lane    = threadIdx.x & 31;
    const int half    = lane >> 4;
    const int l16     = lane & 15;
    const int rowTile = blockIdx.x * 4 + w;
    const int rowbase = rowTile * 16;
    const int b       = rowTile >> 8;

    // ---- load loop-invariant A (16 k-rows x 128) in WMMA A layout ----
    v16h a[4];
    {
        const _Float16* krow = kf + (size_t)(rowbase + l16) * DMODEL;
        for (int c = 0; c < 4; ++c) {
            const int d0 = 32 * c + 16 * half;
            a[c] = *(const v16h*)(krow + d0);
        }
    }

    v8f o[8];
    for (int t = 0; t < 8; ++t) o[t] = (v8f){};
    float run_max[8], run_sum[8];
    for (int r = 0; r < 8; ++r) { run_max[r] = -INFINITY; run_sum[r] = 0.0f; }

    const _Float16* qbatch = qf + (size_t)b * SEQ * DMODEL;
    const _Float16* vbatch = vt + (size_t)b * DMODEL * SEQ;
    _Float16* pl = lds_p[w];
    const float LOG2E = 1.44269504088896340736f;

    for (int j = 0; j < SEQ; j += 32) {
        // ---- S tiles: two 16x16, contraction over d=128 (4 WMMAs each) ----
        v8f s0 = {}, s1 = {};
        for (int c = 0; c < 4; ++c) {
            const int d0 = 32 * c + 16 * half;                 // B layout K = 16*half + e
            v16h b0 = *(const v16h*)(qbatch + (size_t)(j + l16) * DMODEL + d0);
            v16h b1 = *(const v16h*)(qbatch + (size_t)(j + 16 + l16) * DMODEL + d0);
            s0 = wmma_f16(a[c], b0, s0);
            s1 = wmma_f16(a[c], b1, s1);
        }

        // ---- online softmax over the 32 new columns ----
        float corr[8];
        for (int r = 0; r < 8; ++r) {
            float mx = fmaxf(s0[r], s1[r]);
            for (int m = 1; m < 16; m <<= 1) mx = fmaxf(mx, __shfl_xor(mx, m, 32));
            const float nm = fmaxf(run_max[r], mx);
            corr[r] = exp2f((run_max[r] - nm) * LOG2E);
            run_max[r] = nm;
            const float p0 = exp2f((s0[r] - nm) * LOG2E);
            const float p1 = exp2f((s1[r] - nm) * LOG2E);
            float ps = p0 + p1;
            for (int m = 1; m < 16; m <<= 1) ps += __shfl_xor(ps, m, 32);
            run_sum[r] = run_sum[r] * corr[r] + ps;
            const int row = 8 * half + r;                      // C-layout row this lane owns
            pl[row * 32 + l16]      = (_Float16)p0;
            pl[row * 32 + 16 + l16] = (_Float16)p1;
        }
        asm volatile("" ::: "memory");   // LDS ops are in-order per wave (DScnt)

        // ---- reload P (16x32) in A layout from LDS ----
        v16h pa;
        {
            const _Float16* pr = pl + l16 * 32;
            pa = *(const v16h*)(pr + 16 * half);
        }

        // ---- rescale accumulator, then O += P @ V (8 WMMAs) ----
        for (int t = 0; t < 8; ++t)
            for (int r = 0; r < 8; ++r) o[t][r] *= corr[r];

        for (int t = 0; t < 8; ++t) {
            // B[kk][n] = v[j+kk][16t+n] = vt[b][16t+n][j+kk]; contiguous per lane
            v16h bv_ = *(const v16h*)(vbatch + (size_t)(16 * t + l16) * SEQ + j + 16 * half);
            o[t] = wmma_f16(pa, bv_, o[t]);
        }
    }

    // ---- finalize: divide by l, store fp32 ----
    float inv[8];
    for (int r = 0; r < 8; ++r) inv[r] = 1.0f / run_sum[r];
    for (int t = 0; t < 8; ++t)
        for (int r = 0; r < 8; ++r)
            out[(size_t)(rowbase + 8 * half + r) * DMODEL + 16 * t + l16] = o[t][r] * inv[r];
}

// ---------------------------------------------------------------------------
extern "C" void kernel_launch(void* const* d_in, const int* in_sizes, int n_in,
                              void* d_out, int out_size, void* d_ws, size_t ws_size,
                              hipStream_t stream) {
    const float* x  = (const float*)d_in[0];
    const float* Wk = (const float*)d_in[1];
    const float* bk = (const float*)d_in[2];
    const float* Wq = (const float*)d_in[3];
    const float* bq = (const float*)d_in[4];
    const float* Wv = (const float*)d_in[5];
    const float* bv = (const float*)d_in[6];
    float* out = (float*)d_out;

    // workspace: k(f16) | q(f16) | vT(f16)  = 3 * 32768*128*2 B = 24 MB
    _Float16* kf = (_Float16*)d_ws;
    _Float16* qf = kf + (size_t)NROWS * DMODEL;
    _Float16* vt = qf + (size_t)NROWS * DMODEL;

    const int blocks = NROWS / 64;   // 4 waves/block * 16 rows/wave
    proj_qkv_kernel<<<blocks, 128, 0, stream>>>(x, Wk, bk, Wq, bq, Wv, bv, kf, qf, vt);
    attn_kernel<<<blocks, 128, 0, stream>>>(kf, qf, vt, out);
}
